// VQDistortionLoss_42391327212291
// MI455X (gfx1250) — matrix operand order repaired and checked by reading, not compiled, not run on hardware
//
#include <hip/hip_runtime.h>
#include <hip/hip_bf16.h>

typedef __attribute__((ext_vector_type(16))) __bf16 v16bf;
typedef __attribute__((ext_vector_type(8)))  float  v8f;

#define D_CONST 512
#define T_CONST 4096
#define B_CONST 8
#define K_CONST 4096
#define N_CONST (B_CONST * T_CONST)
#define ENC_STRIDE 320
#define ZROW 520   // padded LDS z-row length (ushorts): 1040B, bank-conflict free b128 reads
#define PROW 72    // padded LDS probs-row length (ushorts): 144B

__device__ __forceinline__ unsigned short f2bf(float f) {
  unsigned int u = __float_as_uint(f);
  u += 0x7fffu + ((u >> 16) & 1u);   // round-to-nearest-even
  return (unsigned short)(u >> 16);
}

union V16U { v16bf v; uint4 u[2]; };

// A-fragment (16x32 bf16, M x K): lane m = lane&15 -> row m; half h = lane>>4;
// elements 0..7 -> K = k0+8h+j ; elements 8..15 -> K = k0+16+8h+(j-8)
__device__ __forceinline__ v16bf ldA16(const unsigned short* base, int rowStride,
                                       int row0, int k0, int lane) {
  int m = lane & 15, h = lane >> 4;
  const unsigned short* p = base + (row0 + m) * rowStride + k0 + 8 * h;
  V16U x;
  x.u[0] = *(const uint4*)p;
  x.u[1] = *(const uint4*)(p + 16);
  return x.v;
}

// B-fragment (32x16 bf16, K x N) from row-major [n][k] storage (k contiguous):
// lane n = lane&15 -> column n; elements 0..7 -> K = k0+8h+j ; elements 8..15 -> K = k0+16+8h+(j-8)
__device__ __forceinline__ v16bf ldB16(const unsigned short* base, size_t rowStride,
                                       int col0, int k0, int lane) {
  int n = lane & 15, h = lane >> 4;
  const unsigned short* p = base + (size_t)(col0 + n) * rowStride + k0 + 8 * h;
  V16U x;
  x.u[0] = *(const uint4*)p;
  x.u[1] = *(const uint4*)(p + 16);
  return x.v;
}

__device__ __forceinline__ float rmax16(float x) {
  x = fmaxf(x, __shfl_xor(x, 1));
  x = fmaxf(x, __shfl_xor(x, 2));
  x = fmaxf(x, __shfl_xor(x, 4));
  x = fmaxf(x, __shfl_xor(x, 8));
  return x;
}
__device__ __forceinline__ float rsum16(float x) {
  x += __shfl_xor(x, 1);
  x += __shfl_xor(x, 2);
  x += __shfl_xor(x, 4);
  x += __shfl_xor(x, 8);
  return x;
}

// S block (16x16) for wave (r,cg): rows 16r..16r+15 of z-tile vs codes k0c..k0c+15,
// accumulated over D=512 with 16 bf16 WMMAs.
__device__ __forceinline__ v8f score_tile(const unsigned short* zt, const unsigned short* cbf,
                                          int k0c, int r, int lane) {
  v8f S = {0.f, 0.f, 0.f, 0.f, 0.f, 0.f, 0.f, 0.f};
#pragma unroll
  for (int i = 0; i < 16; ++i) {
    v16bf a = ldA16(zt, ZROW, 16 * r, 32 * i, lane);
    v16bf b = ldB16(cbf, D_CONST, k0c, 32 * i, lane);
    S = __builtin_amdgcn_wmma_f32_16x16x32_bf16(false, a, false, b, (short)0, S, false, false);
  }
  return S;
}

// ---------------- prep kernels ----------------

// z2[n] = sum_d feat[b][d][t]^2  (fp32-exact; coalesced along t across lanes)
__global__ void k_zsq(const float* __restrict__ f, float* __restrict__ z2) {
  int n = blockIdx.x * blockDim.x + threadIdx.x;
  int b = n >> 12, t = n & (T_CONST - 1);
  const float* p = f + (size_t)b * D_CONST * T_CONST + t;
  float a = 0.f;
  for (int d = 0; d < D_CONST; ++d) {
    float v = p[(size_t)d * T_CONST];
    a += v * v;
  }
  z2[n] = a;
}

// transpose [B,D,T] f32 -> [N,D] bf16 via 32x32 LDS tiles
__global__ void k_ztr(const float* __restrict__ f, unsigned short* __restrict__ zbf) {
  __shared__ float tile[32][33];
  int tx = threadIdx.x & 31, ty = threadIdx.x >> 5;
  int t0 = blockIdx.x * 32, d0 = blockIdx.y * 32, b = blockIdx.z;
  const float* fb = f + (size_t)b * D_CONST * T_CONST;
#pragma unroll
  for (int yy = 0; yy < 4; ++yy) {
    int d = d0 + ty + 8 * yy;
    tile[ty + 8 * yy][tx] = fb[(size_t)d * T_CONST + t0 + tx];
  }
  __syncthreads();
#pragma unroll
  for (int yy = 0; yy < 4; ++yy) {
    int t = t0 + ty + 8 * yy;
    zbf[(size_t)(b * T_CONST + t) * D_CONST + d0 + tx] = f2bf(tile[tx][ty + 8 * yy]);
  }
}

// codebook row-major bf16 + c2[k] (fp32, fixed-order reduce)
__global__ void k_crm(const float* __restrict__ cb, unsigned short* __restrict__ cbf,
                      float* __restrict__ c2) {
  __shared__ float red[128];
  int k = blockIdx.x, tid = threadIdx.x;
  float4 v = *(const float4*)(cb + (size_t)k * D_CONST + tid * 4);
  ushort4 o;
  o.x = f2bf(v.x); o.y = f2bf(v.y); o.z = f2bf(v.z); o.w = f2bf(v.w);
  *(ushort4*)(cbf + (size_t)k * D_CONST + tid * 4) = o;
  red[tid] = v.x * v.x + v.y * v.y + v.z * v.z + v.w * v.w;
  __syncthreads();
  for (int st = 64; st > 0; st >>= 1) {
    if (tid < st) red[tid] += red[tid + st];
    __syncthreads();
  }
  if (tid == 0) c2[k] = red[0];
}

// codebook transposed [D,K] bf16 (coalesced writes)
__global__ void k_ctr(const float* __restrict__ cb, unsigned short* __restrict__ cT) {
  int d = blockIdx.x;
  for (int k = threadIdx.x; k < K_CONST; k += 256)
    cT[(size_t)d * K_CONST + k] = f2bf(cb[(size_t)k * D_CONST + d]);
}

// ---------------- fused main kernel: single-pass online softmax ----------------

__global__ __launch_bounds__(256) void k_vq_main(
    const unsigned short* __restrict__ zbf, const float* __restrict__ z2,
    const unsigned short* __restrict__ cbf, const unsigned short* __restrict__ cT,
    const float* __restrict__ c2, const float* __restrict__ cbF,
    const long long* __restrict__ codes, const long long* __restrict__ lens,
    float* __restrict__ partials) {
  __shared__ unsigned short zt[32 * ZROW];        // 33,280 B z tile (bf16)
  __shared__ unsigned short probs[2][32 * PROW];  // 9,216 B double-buffered probs (bf16)
  __shared__ float z2s[32];
  __shared__ float wmb[4][32];                    // per-(cg) tile row maxes
  __shared__ float wsb[4][32];                    // per-(cg) final sum-exp partials
  __shared__ float srinv[32];
  __shared__ float rowpart[4][32];
  __shared__ float rowloss[32];

  const int tid = threadIdx.x;
  const int lane = tid & 31;
  const int w = tid >> 5;
  const int cg = w & 3;   // code-column group / D-slice (0..3)
  const int r = w >> 2;   // row group (0..1)
  const int h = lane >> 4;
  const int n0 = blockIdx.x * 32;

  // stage z tile: 32 rows x 512 bf16 (read from LDS by WMMA A-fragments; compiler
  // hoists the 16 loop-invariant A-fragments into VGPRs)
#pragma unroll
  for (int i = 0; i < 8; ++i) {
    int e = (tid + 256 * i) * 8;       // ushort index into 32x512
    int row = e >> 9, col = e & 511;
    *(uint4*)(&zt[row * ZROW + col]) = *(const uint4*)(zbf + (size_t)(n0 + row) * D_CONST + col);
  }
  if (tid < 32) z2s[tid] = z2[n0 + tid];
  __syncthreads();

  float vz2[8];
#pragma unroll
  for (int v = 0; v < 8; ++v) vz2[v] = z2s[16 * r + v + 8 * h];

  // running state: vm = workgroup-consistent running row max, vs = running sum-exp
  // (per-cg partial over this wave's code columns), O = unnormalized recon accumulator
  float vm[8], vs[8];
#pragma unroll
  for (int v = 0; v < 8; ++v) { vm[v] = -3.0e38f; vs[v] = 0.f; }

  v8f zero8 = {0.f, 0.f, 0.f, 0.f, 0.f, 0.f, 0.f, 0.f};
  v8f O[8];
#pragma unroll
  for (int blk = 0; blk < 8; ++blk) O[blk] = zero8;

  int it = 0;
  for (int kb = 0; kb < K_CONST; kb += 64, ++it) {
    int k0c = kb + 16 * cg;

    // prefetch next tile's score-B codebook rows into WGP$ (global_prefetch_b8)
    if (kb + 64 < K_CONST && h == 0) {
      const unsigned short* np = cbf + (size_t)(k0c + 64 + (lane & 15)) * D_CONST;
      __builtin_prefetch(np, 0, 1);
      __builtin_prefetch(np + 128, 0, 1);
      __builtin_prefetch(np + 256, 0, 1);
      __builtin_prefetch(np + 384, 0, 1);
    }

    v8f S = score_tile(zt, cbf, k0c, r, lane);   // 16 WMMA
    float c2k = c2[k0c + (lane & 15)];

    // logits and per-wave per-row max over this wave's 16 codes
    float l[8];
#pragma unroll
    for (int v = 0; v < 8; ++v) {
      float d2 = vz2[v] + c2k - 2.f * S[v];
      l[v] = -sqrtf(fmaxf(d2, 1e-12f));
      float tm = rmax16(l[v]);
      if ((lane & 15) == 0) wmb[cg][16 * r + v + 8 * h] = tm;
    }
    __syncthreads();   // barrier 1: tile maxes visible

    // workgroup-consistent new running max + rescale factor (computed redundantly per wave)
    float scale[8];
#pragma unroll
    for (int v = 0; v < 8; ++v) {
      int row = 16 * r + v + 8 * h;
      float tm = fmaxf(fmaxf(wmb[0][row], wmb[1][row]), fmaxf(wmb[2][row], wmb[3][row]));
      float nm = fmaxf(vm[v], tm);
      scale[v] = __expf(vm[v] - nm);
      vm[v] = nm;
    }

    // unnormalized probs (consistent scaling across all cg waves) -> bf16 LDS
    unsigned short* pb = &probs[it & 1][0];
#pragma unroll
    for (int v = 0; v < 8; ++v) {
      float e = __expf(l[v] - vm[v]);
      vs[v] = vs[v] * scale[v] + rsum16(e);
      pb[(16 * r + v + 8 * h) * PROW + 16 * cg + (lane & 15)] = f2bf(e);
    }
    __syncthreads();   // barrier 2: probs tile ready

    // rescale accumulator rows, then accumulate O += e @ C  (16 WMMA)
#pragma unroll
    for (int blk = 0; blk < 8; ++blk) {
#pragma unroll
      for (int v = 0; v < 8; ++v) O[blk][v] *= scale[v];
    }
#pragma unroll
    for (int q = 0; q < 2; ++q) {
      v16bf a = ldA16(pb, PROW, 16 * r, 32 * q, lane);
#pragma unroll
      for (int blk = 0; blk < 8; ++blk) {
        v16bf b = ldB16(cT, K_CONST, 128 * cg + 16 * blk, kb + 32 * q, lane);
        O[blk] = __builtin_amdgcn_wmma_f32_16x16x32_bf16(false, a, false, b, (short)0, O[blk], false, false);
      }
    }
  }

  // combine sum-exp across the 4 cg waves -> per-row normalizer
  if ((lane & 15) == 0) {
#pragma unroll
    for (int v = 0; v < 8; ++v) wsb[cg][16 * r + v + 8 * h] = vs[v];
  }
  __syncthreads();
  if (tid < 32) {
    float s = wsb[0][tid] + wsb[1][tid] + wsb[2][tid] + wsb[3][tid];
    srinv[tid] = 1.0f / s;
  }
  __syncthreads();
  float vsi[8];
#pragma unroll
  for (int v = 0; v < 8; ++v) vsi[v] = srinv[16 * r + v + 8 * h];

  // ---- epilogue: masked MSE vs teacher embedding (fp32 codebook) ----
  float racc[8];
#pragma unroll
  for (int v = 0; v < 8; ++v) racc[v] = 0.f;
#pragma unroll
  for (int v = 0; v < 8; ++v) {
    int n = n0 + 16 * r + v + 8 * h;
    long long code = codes[n];
    const float* tp = cbF + (size_t)code * D_CONST + 128 * cg + (lane & 15);
#pragma unroll
    for (int blk = 0; blk < 8; ++blk) {
      float t = tp[16 * blk];
      float df = O[blk][v] * vsi[v] - t;
      racc[v] += df * df;
    }
  }
#pragma unroll
  for (int v = 0; v < 8; ++v) {
    float s = rsum16(racc[v]);
    if ((lane & 15) == 0) rowpart[cg][16 * r + v + 8 * h] = s;
  }
  __syncthreads();
  if (tid < 32) {
    float tot = rowpart[0][tid] + rowpart[1][tid] + rowpart[2][tid] + rowpart[3][tid];
    float mean = tot * (1.0f / D_CONST);
    int n = n0 + tid;
    int b = n >> 12;
    int t = n & (T_CONST - 1);
    float msk = ((long long)t * ENC_STRIDE < lens[b]) ? 1.0f : 0.0f;
    rowloss[tid] = mean * msk;
  }
  __syncthreads();
  if (tid == 0) {
    float a = 0.f;
#pragma unroll
    for (int i = 0; i < 32; ++i) a += rowloss[i];
    partials[blockIdx.x] = a;
  }
}

// fixed-order final reduction (deterministic, no float atomics)
__global__ void k_final(const float* __restrict__ partials, const long long* __restrict__ lens,
                        float* __restrict__ out) {
  __shared__ long long cnts[256];
  int tid = threadIdx.x;
  long long cnt = 0;
  for (int n = tid; n < N_CONST; n += 256) {
    int b = n >> 12;
    int t = n & (T_CONST - 1);
    if ((long long)t * ENC_STRIDE < lens[b]) cnt++;
  }
  cnts[tid] = cnt;
  __syncthreads();
  if (tid == 0) {
    long long c = 0;
    for (int i = 0; i < 256; ++i) c += cnts[i];
    float num = 0.f;
    for (int i = 0; i < N_CONST / 32; ++i) num += partials[i];
    out[0] = num / ((float)c + 1e-8f);
  }
}

extern "C" void kernel_launch(void* const* d_in, const int* in_sizes, int n_in,
                              void* d_out, int out_size, void* d_ws, size_t ws_size,
                              hipStream_t stream) {
  (void)in_sizes; (void)n_in; (void)out_size; (void)ws_size;
  const float* feat = (const float*)d_in[0];        // [B, D, T] f32
  const float* cb = (const float*)d_in[1];          // [K, D] f32
  const long long* codes = (const long long*)d_in[2];  // [B, T] i64
  const long long* lens = (const long long*)d_in[3];   // [B] i64
  float* out = (float*)d_out;
  char* ws = (char*)d_ws;

  size_t off = 0;
  unsigned short* zbf = (unsigned short*)(ws + off); off += (size_t)N_CONST * D_CONST * 2;  // 32 MB
  float* z2 = (float*)(ws + off);                   off += (size_t)N_CONST * 4;             // 128 KB
  unsigned short* cbf = (unsigned short*)(ws + off); off += (size_t)K_CONST * D_CONST * 2;  // 4 MB
  unsigned short* cT = (unsigned short*)(ws + off);  off += (size_t)D_CONST * K_CONST * 2;  // 4 MB
  float* c2 = (float*)(ws + off);                   off += (size_t)K_CONST * 4;             // 16 KB
  float* partials = (float*)(ws + off);             off += (size_t)(N_CONST / 32) * 4;      // 4 KB

  k_zsq<<<N_CONST / 256, 256, 0, stream>>>(feat, z2);
  k_ztr<<<dim3(T_CONST / 32, D_CONST / 32, B_CONST), 256, 0, stream>>>(feat, zbf);
  k_crm<<<K_CONST, 128, 0, stream>>>(cb, cbf, c2);
  k_ctr<<<D_CONST, 256, 0, stream>>>(cb, cT);
  k_vq_main<<<N_CONST / 32, 256, 0, stream>>>(zbf, z2, cbf, cT, c2, cb, codes, lens, partials);
  k_final<<<1, 256, 0, stream>>>(partials, lens, out);
}
